// InterBlock_48069273977224
// MI455X (gfx1250) — matrix operand order repaired and checked by reading, not compiled, not run on hardware
//
#include <hip/hip_runtime.h>

typedef __attribute__((ext_vector_type(16))) _Float16 v16h;
typedef __attribute__((ext_vector_type(8)))  float    v8f;

namespace {

constexpr int   kHidden       = 64;
constexpr int   kRbf          = 300;
constexpr int   kKpad         = 320;   // 300 padded to a multiple of 32
constexpr int   kWaves        = 8;     // waves per block (256 threads, wave32)
constexpr float kCutoff       = 5.0f;
constexpr float kGamma        = 10.0f;
constexpr float kPiOverCutoff = 0.62831853071795864769f; // pi / 5

__device__ __forceinline__ v8f wmma16(v16h a, v16h b, v8f c) {
  // D = A(16x32 f16) * B(32x16 f16) + C(16x16 f32)
  return __builtin_amdgcn_wmma_f32_16x16x32_f16(
      /*neg_a=*/false, a, /*neg_b=*/false, b,
      /*c_mod=*/(short)0, c, /*reuse_a=*/false, /*reuse_b=*/false);
}

__device__ __forceinline__ float fast_tanh(float x) {
  float cx = fminf(fmaxf(x, -9.0f), 9.0f);
  float e  = __expf(2.0f * cx);
  return (e - 1.0f) / (e + 1.0f);
}

// ---- CDNA5 async global->LDS staging (gfx1250 ASYNCcnt path) ----
// Per-lane 16B copy. LDS byte offset = low 32 bits of generic shared pointer.
__device__ __forceinline__ void async_copy_16B(const void* g, void* l) {
  uint32_t loff  = (uint32_t)(uintptr_t)l;
  uint64_t gaddr = (uint64_t)(uintptr_t)g;
  asm volatile("global_load_async_to_lds_b128 %0, %1, off"
               :
               : "v"(loff), "v"(gaddr)
               : "memory");
}
__device__ __forceinline__ void wait_async_lds() {
  asm volatile("s_wait_asynccnt 0" ::: "memory");
}
// In-wave LDS store->load fence (LDS is in-order per wave; this pins the
// compiler's ordering and drains DScnt).
__device__ __forceinline__ void wait_ds0() {
  asm volatile("s_wait_dscnt 0" ::: "memory");
}

// A-tile (16 rows x 32 K) in CDNA5 16-bit A layout, from f32 row-major.
// lanes 0-15  (row=lane):    elems 0..7 -> K=kc+0..7,  8..15 -> K=kc+16..23
// lanes 16-31 (row=lane-16): elems 0..7 -> K=kc+8..15, 8..15 -> K=kc+24..31
__device__ __forceinline__ v16h load_a_f32(const float* base, int stride,
                                           int lane, int kc, int mclamp) {
  int m  = lane & 15;
  if (m > mclamp) m = mclamp;
  const int hi = (lane >> 4) & 1;
  const float* p = base + (size_t)m * stride + kc + hi * 8;
  v16h a;
#pragma unroll
  for (int e = 0; e < 8; ++e) a[e] = (_Float16)p[e];
#pragma unroll
  for (int e = 0; e < 8; ++e) a[8 + e] = (_Float16)p[16 + e];
  return a;
}

// Same A layout, f16 source (LDS-resident intermediates).
__device__ __forceinline__ v16h load_a_f16(const _Float16* base, int stride,
                                           int lane, int kc) {
  const int m  = lane & 15;
  const int hi = (lane >> 4) & 1;
  const _Float16* p = base + (size_t)m * stride + kc + hi * 8;
  v16h a;
#pragma unroll
  for (int e = 0; e < 8; ++e) a[e] = p[e];
#pragma unroll
  for (int e = 0; e < 8; ++e) a[8 + e] = p[16 + e];
  return a;
}

// B-tile (32 K x 16 cols). B[k][n] = Wt[ncol0+n][k] with Wt row-major
// [64][kpad] (computes X @ Wt^T). lanes 0-15: col n, elems 0..7 -> K=kc+0..7,
// 8..15 -> K=kc+16..23; lanes 16-31: K=kc+8..15, kc+24..31.
__device__ __forceinline__ v16h load_b_f16(const _Float16* Wt, int kpad,
                                           int lane, int ncol0, int kc) {
  const int n  = lane & 15;
  const int hi = (lane >> 4) & 1;
  const _Float16* p = Wt + (size_t)(ncol0 + n) * kpad + kc + hi * 8;
  v16h b;
#pragma unroll
  for (int e = 0; e < 8; ++e) b[e] = p[e];
#pragma unroll
  for (int e = 0; e < 8; ++e) b[8 + e] = p[16 + e];
  return b;
}

} // namespace

// ---------------- setup kernels ----------------

__global__ void convert_weights_kernel(const float* __restrict__ W1,
                                       const float* __restrict__ W2,
                                       const float* __restrict__ Waw,
                                       _Float16* __restrict__ W1h,
                                       _Float16* __restrict__ W2h,
                                       _Float16* __restrict__ Wawh) {
  int t = blockIdx.x * blockDim.x + threadIdx.x;
  if (t < kHidden * kKpad) {
    int row = t / kKpad, col = t - row * kKpad;
    W1h[t] = (_Float16)((col < kRbf) ? W1[row * kRbf + col] : 0.0f);
  }
  if (t < kHidden * kHidden) {
    W2h[t]  = (_Float16)W2[t];
    Wawh[t] = (_Float16)Waw[t];
  }
}

__global__ void zero_kernel(float* __restrict__ p, int nel) {
  int t = blockIdx.x * blockDim.x + threadIdx.x;
  if (t < nel) p[t] = 0.0f;
}

// ---------------- h = x @ W_aw^T + b_aw ----------------
// 8 waves/block; W_aw (f16) staged in LDS via async copy; each wave handles
// one 16-row tile per grid-stride step: 2 K-chunks x 4 N-tiles = 8 WMMA.

__global__ void __launch_bounds__(256)
node_linear_kernel(const float* __restrict__ x,
                   const _Float16* __restrict__ Wawh,
                   const float* __restrict__ baw,
                   float* __restrict__ hout, int nNodes) {
  __shared__ _Float16 s_w[kHidden * kHidden];

  const int tid = threadIdx.x;
  for (int i = tid; i < (kHidden * kHidden) / 8; i += blockDim.x)
    async_copy_16B(Wawh + i * 8, &s_w[i * 8]);
  wait_async_lds();
  __syncthreads();

  const int wave = tid >> 5;
  const int lane = tid & 31;
  const int n    = lane & 15;
  const int hi   = (lane >> 4) & 1;
  const int nTiles = (nNodes + 15) / 16;

  // loop-invariant per-lane biases (hoisted out of the persistent loop)
  float bias4[4];
#pragma unroll
  for (int nt = 0; nt < 4; ++nt) bias4[nt] = baw[nt * 16 + n];

  for (int tile = blockIdx.x * kWaves + wave; tile < nTiles;
       tile += gridDim.x * kWaves) {
    const int  row0 = tile * 16;
    const bool full = (row0 + 16 <= nNodes);
    const int  mclamp = full ? 15 : (nNodes - row0 - 1);

    v8f acc[4] = {};
#pragma unroll
    for (int kc = 0; kc < kHidden; kc += 32) {
      v16h a = load_a_f32(x + (size_t)row0 * kHidden, kHidden, lane, kc, mclamp);
#pragma unroll
      for (int nt = 0; nt < 4; ++nt) {
        v16h b = load_b_f16(s_w, kHidden, lane, nt * 16, kc);
        acc[nt] = wmma16(a, b, acc[nt]);
      }
    }
    if (full) {
      // uniform fast path: no per-element EXEC masking
      float* op = hout + (size_t)(row0 + hi * 8) * kHidden + n;
#pragma unroll
      for (int nt = 0; nt < 4; ++nt) {
#pragma unroll
        for (int g = 0; g < 8; ++g)
          op[(size_t)g * kHidden + nt * 16] = acc[nt][g] + bias4[nt];
      }
    } else {
#pragma unroll
      for (int nt = 0; nt < 4; ++nt) {
#pragma unroll
        for (int g = 0; g < 8; ++g) {
          int row = row0 + g + hi * 8;
          if (row < nNodes)
            hout[(size_t)row * kHidden + nt * 16 + n] = acc[nt][g] + bias4[nt];
        }
      }
    }
  }
}

// ---------------- edge interaction (dominant kernel) ----------------
// W1h (40KB) + W2h (8KB) staged once per block into LDS (async path),
// amortized over a grid-stride loop: each wave owns a 16-edge tile.
// RBF generated in registers directly in WMMA A layout; layer1 = 40 WMMA,
// layer2 = 8 WMMA per tile; then fused cutoff*tanh, h[src] gather and
// atomic scatter-add into agg[dst].

__global__ void __launch_bounds__(256)
edge_interaction_kernel(const float* __restrict__ r,
                        const int* __restrict__ src,
                        const int* __restrict__ dst,
                        const float* __restrict__ h,
                        const _Float16* __restrict__ W1h,
                        const float* __restrict__ b1,
                        const _Float16* __restrict__ W2h,
                        const float* __restrict__ b2,
                        float* __restrict__ agg,
                        int nEdges) {
  __shared__ _Float16 s_w1[kHidden * kKpad];
  __shared__ _Float16 s_w2[kHidden * kHidden];
  __shared__ _Float16 s_t1[kWaves][16 * kHidden];
  __shared__ int      s_src[kWaves][16];
  __shared__ int      s_dst[kWaves][16];
  __shared__ float    s_cut[kWaves][16];

  const int tid = threadIdx.x;
  for (int i = tid; i < (kHidden * kKpad) / 8; i += blockDim.x)
    async_copy_16B(W1h + i * 8, &s_w1[i * 8]);
  for (int i = tid; i < (kHidden * kHidden) / 8; i += blockDim.x)
    async_copy_16B(W2h + i * 8, &s_w2[i * 8]);
  wait_async_lds();
  __syncthreads();

  const int wave = tid >> 5;
  const int lane = tid & 31;
  const int n    = lane & 15;
  const int hi   = (lane >> 4) & 1;
  const int nTiles = (nEdges + 15) / 16;
  const int strideE = 16 * gridDim.x * kWaves;

  // loop-invariant per-lane biases (hoisted out of the persistent loop)
  float bias1[4], bias2[4];
#pragma unroll
  for (int nt = 0; nt < 4; ++nt) {
    bias1[nt] = b1[nt * 16 + n];
    bias2[nt] = b2[nt * 16 + n];
  }

  for (int tile = blockIdx.x * kWaves + wave; tile < nTiles;
       tile += gridDim.x * kWaves) {
    const int  ebase = tile * 16;
    const bool full  = (ebase + 16 <= nEdges);
    __builtin_prefetch(r + ebase + strideE, 0, 0);

    if (lane < 16) {
      int e = ebase + lane;
      if (full || e < nEdges) {
        s_src[wave][lane] = src[e];
        s_dst[wave][lane] = dst[e];
        float rr = r[e];
        float c  = 0.5f * (__cosf(rr * kPiOverCutoff) + 1.0f);
        s_cut[wave][lane] = (rr < kCutoff) ? c : 0.0f;
      } else {
        s_src[wave][lane] = 0; s_dst[wave][lane] = 0; s_cut[wave][lane] = 0.0f;
      }
    }
    wait_ds0();  // in-wave LDS ordering

    const int em = ebase + n;
    const float rm = (full || em < nEdges) ? r[em] : 0.0f;

    // ---- layer 1: rbf(E x 320) @ W1^T  (B from LDS) ----
    v8f acc[4] = {};
    for (int kc = 0; kc < kKpad; kc += 32) {
      v16h a;
      const int kb = kc + hi * 8;
#pragma unroll
      for (int e2 = 0; e2 < 8; ++e2) {
        int   k = kb + e2;
        float d = rm - 0.1f * (float)k;
        float v = (k < kRbf) ? __expf(-kGamma * d * d) : 0.0f;
        a[e2] = (_Float16)v;
      }
#pragma unroll
      for (int e2 = 0; e2 < 8; ++e2) {
        int   k = kb + 16 + e2;
        float d = rm - 0.1f * (float)k;
        float v = (k < kRbf) ? __expf(-kGamma * d * d) : 0.0f;
        a[8 + e2] = (_Float16)v;
      }
#pragma unroll
      for (int nt = 0; nt < 4; ++nt) {
        v16h b = load_b_f16(s_w1, kKpad, lane, nt * 16, kc);
        acc[nt] = wmma16(a, b, acc[nt]);
      }
    }

    // tanh(.+b1) -> per-wave LDS (C-layout -> A-layout transpose)
#pragma unroll
    for (int nt = 0; nt < 4; ++nt) {
#pragma unroll
      for (int g = 0; g < 8; ++g)
        s_t1[wave][(g + hi * 8) * kHidden + nt * 16 + n] =
            (_Float16)fast_tanh(acc[nt][g] + bias1[nt]);
    }
    wait_ds0();  // in-wave LDS ordering

    // ---- layer 2: t1(E x 64) @ W2^T  (A and B from LDS) ----
    v8f acc2[4] = {};
#pragma unroll
    for (int kc = 0; kc < kHidden; kc += 32) {
      v16h a = load_a_f16(s_t1[wave], kHidden, lane, kc);
#pragma unroll
      for (int nt = 0; nt < 4; ++nt) {
        v16h b = load_b_f16(s_w2, kHidden, lane, nt * 16, kc);
        acc2[nt] = wmma16(a, b, acc2[nt]);
      }
    }

    // ---- W = tanh(.+b2)*cutoff; msg = h[src]*W; scatter-add at dst ----
    if (full) {
      // uniform fast path: no per-element EXEC masking
#pragma unroll
      for (int nt = 0; nt < 4; ++nt) {
        int hidx = nt * 16 + n;
#pragma unroll
        for (int g = 0; g < 8; ++g) {
          int   ml  = g + hi * 8;
          float w   = fast_tanh(acc2[nt][g] + bias2[nt]) * s_cut[wave][ml];
          float msg = h[(size_t)s_src[wave][ml] * kHidden + hidx] * w;
          atomicAdd(&agg[(size_t)s_dst[wave][ml] * kHidden + hidx], msg);
        }
      }
    } else {
#pragma unroll
      for (int nt = 0; nt < 4; ++nt) {
        int hidx = nt * 16 + n;
#pragma unroll
        for (int g = 0; g < 8; ++g) {
          int ml = g + hi * 8;
          if (ebase + ml < nEdges) {
            float w   = fast_tanh(acc2[nt][g] + bias2[nt]) * s_cut[wave][ml];
            float msg = h[(size_t)s_src[wave][ml] * kHidden + hidx] * w;
            atomicAdd(&agg[(size_t)s_dst[wave][ml] * kHidden + hidx], msg);
          }
        }
      }
    }
  }
}

// ---------------- out = tanh(agg@W^T+b) @ W^T + b ----------------

__global__ void __launch_bounds__(256)
node_output_kernel(const float* __restrict__ agg,
                   const _Float16* __restrict__ Wawh,
                   const float* __restrict__ baw,
                   float* __restrict__ out, int nNodes) {
  __shared__ _Float16 s_w[kHidden * kHidden];
  __shared__ _Float16 s_t[kWaves][16 * kHidden];

  const int tid = threadIdx.x;
  for (int i = tid; i < (kHidden * kHidden) / 8; i += blockDim.x)
    async_copy_16B(Wawh + i * 8, &s_w[i * 8]);
  wait_async_lds();
  __syncthreads();

  const int wave = tid >> 5;
  const int lane = tid & 31;
  const int n    = lane & 15;
  const int hi   = (lane >> 4) & 1;
  const int nTiles = (nNodes + 15) / 16;

  // loop-invariant per-lane biases (hoisted out of the persistent loop)
  float bias4[4];
#pragma unroll
  for (int nt = 0; nt < 4; ++nt) bias4[nt] = baw[nt * 16 + n];

  for (int tile = blockIdx.x * kWaves + wave; tile < nTiles;
       tile += gridDim.x * kWaves) {
    const int  row0 = tile * 16;
    const bool full = (row0 + 16 <= nNodes);
    const int  mclamp = full ? 15 : (nNodes - row0 - 1);

    v8f acc[4] = {};
#pragma unroll
    for (int kc = 0; kc < kHidden; kc += 32) {
      v16h a = load_a_f32(agg + (size_t)row0 * kHidden, kHidden, lane, kc, mclamp);
#pragma unroll
      for (int nt = 0; nt < 4; ++nt) {
        v16h b = load_b_f16(s_w, kHidden, lane, nt * 16, kc);
        acc[nt] = wmma16(a, b, acc[nt]);
      }
    }
#pragma unroll
    for (int nt = 0; nt < 4; ++nt) {
#pragma unroll
      for (int g = 0; g < 8; ++g)
        s_t[wave][(g + hi * 8) * kHidden + nt * 16 + n] =
            (_Float16)fast_tanh(acc[nt][g] + bias4[nt]);
    }
    wait_ds0();  // in-wave LDS ordering

    v8f acc2[4] = {};
#pragma unroll
    for (int kc = 0; kc < kHidden; kc += 32) {
      v16h a = load_a_f16(s_t[wave], kHidden, lane, kc);
#pragma unroll
      for (int nt = 0; nt < 4; ++nt) {
        v16h b = load_b_f16(s_w, kHidden, lane, nt * 16, kc);
        acc2[nt] = wmma16(a, b, acc2[nt]);
      }
    }
    if (full) {
      // uniform fast path: no per-element EXEC masking
      float* op = out + (size_t)(row0 + hi * 8) * kHidden + n;
#pragma unroll
      for (int nt = 0; nt < 4; ++nt) {
#pragma unroll
        for (int g = 0; g < 8; ++g)
          op[(size_t)g * kHidden + nt * 16] = acc2[nt][g] + bias4[nt];
      }
    } else {
#pragma unroll
      for (int nt = 0; nt < 4; ++nt) {
#pragma unroll
        for (int g = 0; g < 8; ++g) {
          int row = row0 + g + hi * 8;
          if (row < nNodes)
            out[(size_t)row * kHidden + nt * 16 + n] = acc2[nt][g] + bias4[nt];
        }
      }
    }
  }
}

// ---------------- host launcher ----------------

extern "C" void kernel_launch(void* const* d_in, const int* in_sizes, int n_in,
                              void* d_out, int out_size, void* d_ws, size_t ws_size,
                              hipStream_t stream) {
  const float* x   = (const float*)d_in[0];
  const float* r   = (const float*)d_in[1];
  const int*   ei  = (const int*)  d_in[2];
  const float* W1  = (const float*)d_in[3];
  const float* b1  = (const float*)d_in[4];
  const float* W2  = (const float*)d_in[5];
  const float* b2  = (const float*)d_in[6];
  const float* Waw = (const float*)d_in[7];
  const float* baw = (const float*)d_in[8];

  const int nNodes = in_sizes[0] / kHidden;   // 50000
  const int nEdges = in_sizes[1];             // 1000000
  const int* src = ei;                        // edge_index[0]
  const int* dst = ei + nEdges;               // edge_index[1]

  // workspace layout
  char* ws = (char*)d_ws;
  const size_t aggBytes = (size_t)nNodes * kHidden * sizeof(float);
  float*     agg  = (float*)ws;
  float*     hbuf = (float*)(ws + aggBytes);
  _Float16*  W1h  = (_Float16*)(ws + 2 * aggBytes);
  _Float16*  W2h  = W1h + (size_t)kHidden * kKpad;
  _Float16*  Wawh = W2h + (size_t)kHidden * kHidden;
  (void)ws_size; (void)n_in; (void)out_size;

  const int nodeElems = nNodes * kHidden;
  const int nodeTiles = (nNodes + 15) / 16;
  const int edgeTiles = (nEdges + 15) / 16;

  int egrid = (edgeTiles + kWaves - 1) / kWaves;
  if (egrid > 1024) egrid = 1024;            // persistent-style, amortize staging
  int ngrid = (nodeTiles + kWaves - 1) / kWaves;
  if (ngrid > 512) ngrid = 512;

  convert_weights_kernel<<<(kHidden * kKpad + 255) / 256, 256, 0, stream>>>(
      W1, W2, Waw, W1h, W2h, Wawh);
  zero_kernel<<<(nodeElems + 255) / 256, 256, 0, stream>>>(agg, nodeElems);
  node_linear_kernel<<<ngrid, 256, 0, stream>>>(x, Wawh, baw, hbuf, nNodes);
  edge_interaction_kernel<<<egrid, 256, 0, stream>>>(
      r, src, dst, hbuf, W1h, b1, W2h, b2, agg, nEdges);
  node_output_kernel<<<ngrid, 256, 0, stream>>>(agg, Wawh, baw,
                                                (float*)d_out, nNodes);
}
